// BalancingLoss_87883620811481
// MI455X (gfx1250) — compile-verified, matrix operand order checked by reading, not run on hardware
//
#include <hip/hip_runtime.h>
#include <hip/hip_bf16.h>

// gfx1250 async global->LDS copy path, guarded so the file always compiles.
#if defined(__has_builtin)
#  if __has_builtin(__builtin_amdgcn_global_load_async_to_lds_b128) && \
      __has_builtin(__builtin_amdgcn_s_wait_asynccnt)
#    define USE_ASYNC 1
#  endif
#endif
#ifndef USE_ASYNC
#  define USE_ASYNC 0
#endif

typedef __attribute__((ext_vector_type(16))) _Float16 v16h;
typedef __attribute__((ext_vector_type(8)))  float    v8f;
typedef __attribute__((__vector_size__(4 * sizeof(int)))) int v4i;  // b128 payload

#define NEXP   64                 // n_routed_experts
#define KSEL   8                  // num_experts_per_tok
#define LAYERS 32
#define TOKENS 16384
#define BPL    16                 // blocks per layer
#define TPB    (TOKENS / BPL)     // tokens per block = 1024
#define THREADS 256               // 8 wave32 per block
#define ITERS  (TPB / THREADS)    // 4 tokens per lane
#define ROWB   272                // padded LDS row stride: 256B row + 16B pad
#define WAVE_LDS (32 * ROWB)      // 8704 B staging per wave

// ---------------------------------------------------------------------------
// Stage 0: zero the [L,64] accumulator arrays in workspace (harness poisons ws)
// ---------------------------------------------------------------------------
__global__ void bal_zero(float* __restrict__ p, int n) {
    int i = blockIdx.x * blockDim.x + threadIdx.x;
    if (i < n) p[i] = 0.0f;
}

// ---------------------------------------------------------------------------
// Stage 1: stream logits once; per-token softmax + top-8; accumulate
//   gW[l][e] = sum_t softmax(l,t)[e]      (for rw_mean)
//   gC[l][e] = #tokens with e in top-8    (tokens_per_expert)
// Lane = token. Global->LDS staging is fully coalesced (512B/instruction) via
// async-to-LDS; per-lane row consumption from padded LDS rows.
// ---------------------------------------------------------------------------
__global__ __launch_bounds__(THREADS) void bal_stage1(const float* __restrict__ logits,
                                                      float* __restrict__ gW,
                                                      float* __restrict__ gC) {
    const int layer = blockIdx.x / BPL;
    const int blk   = blockIdx.x % BPL;
    const int tid   = threadIdx.x;
    const int lane  = tid & 31;

#if USE_ASYNC
    __shared__ __align__(16) char smem[8 * WAVE_LDS];
    const int waveId = tid >> 5;
    char* lbase = smem + waveId * WAVE_LDS;
    // producer lane offset: chunk j of 512B covers rows (2j, 2j+1);
    // lane ln writes 16B at row (2j + (ln>>4)), col (ln&15)*16
    const int prodOff = (lane >> 4) * ROWB + (lane & 15) * 16;
#endif

    float    accW[NEXP];
    unsigned accC[NEXP / 2];          // two 16-bit counters per register
    #pragma unroll
    for (int i = 0; i < NEXP; ++i) accW[i] = 0.0f;
    #pragma unroll
    for (int i = 0; i < NEXP / 2; ++i) accC[i] = 0u;

    const float* layerBase = logits + (size_t)layer * TOKENS * NEXP;

    for (int it = 0; it < ITERS; ++it) {
        const int tokBase = blk * TPB + it * THREADS;
        if (it + 1 < ITERS)
            __builtin_prefetch(layerBase + (size_t)(tokBase + THREADS + tid) * NEXP, 0, 0);

        float v[NEXP];
#if USE_ASYNC
        // ---- coalesced async stage of this wave's 32 rows (8 KB) into LDS ----
        const char* gsrc = (const char*)(layerBase +
                           (size_t)(tokBase + (waveId << 5)) * NEXP);
        #pragma unroll
        for (int j = 0; j < 16; ++j) {
            __builtin_amdgcn_global_load_async_to_lds_b128(
                (__attribute__((address_space(1))) v4i*)
                    (gsrc + (size_t)j * 512 + lane * 16),
                (__attribute__((address_space(3))) v4i*)
                    (lbase + j * (2 * ROWB) + prodOff),
                0, 0);
        }
        __builtin_amdgcn_s_wait_asynccnt(0);
        // ---- per-lane row consumption from LDS (2-way conflict max) ----
        const char* lrow = lbase + lane * ROWB;
        #pragma unroll
        for (int k = 0; k < NEXP / 4; ++k) {
            float4 q = *(const float4*)(lrow + k * 16);
            v[4 * k + 0] = q.x; v[4 * k + 1] = q.y;
            v[4 * k + 2] = q.z; v[4 * k + 3] = q.w;
        }
#else
        const float4* row = (const float4*)(layerBase + (size_t)(tokBase + tid) * NEXP);
        #pragma unroll
        for (int k = 0; k < NEXP / 4; ++k) {
            float4 q = row[k];
            v[4 * k + 0] = q.x; v[4 * k + 1] = q.y;
            v[4 * k + 2] = q.z; v[4 * k + 3] = q.w;
        }
#endif

        float m = v[0];
        #pragma unroll
        for (int i = 1; i < NEXP; ++i) m = fmaxf(m, v[i]);

        float s = 0.0f;
        #pragma unroll
        for (int i = 0; i < NEXP; ++i) { v[i] = __expf(v[i] - m); s += v[i]; }

        // branchless top-8 insertion (descending t[0..7]); all v[i] > 0;
        // selection on exp values == selection on logits (softmax monotonic)
        float t[KSEL];
        #pragma unroll
        for (int r = 0; r < KSEL; ++r) t[r] = -1.0f;
        #pragma unroll
        for (int i = 0; i < NEXP; ++i) {
            float y = v[i];
            #pragma unroll
            for (int r = 0; r < KSEL; ++r) {
                float hi = fmaxf(t[r], y);
                y = fminf(t[r], y);
                t[r] = hi;
            }
        }
        const float thr  = t[KSEL - 1];   // 8th largest (ties: measure-zero)
        const float rinv = 1.0f / s;

        #pragma unroll
        for (int i = 0; i < NEXP / 2; ++i) {
            accW[i]      += v[i] * rinv;
            accW[i + 32] += v[i + 32] * rinv;
            accC[i] += (v[i] >= thr ? 1u : 0u) + (v[i + 32] >= thr ? 0x10000u : 0u);
        }
    }

    // wave32 cross-lane reduction (butterfly)
    #pragma unroll
    for (int i = 0; i < NEXP; ++i) {
        #pragma unroll
        for (int d = 1; d < 32; d <<= 1) accW[i] += __shfl_xor(accW[i], d, 32);
    }
    #pragma unroll
    for (int i = 0; i < NEXP / 2; ++i) {
        #pragma unroll
        for (int d = 1; d < 32; d <<= 1) accC[i] += __shfl_xor(accC[i], d, 32);
    }

    // one active lane per value -> global f32 atomics into [L,64] scratch
    #pragma unroll
    for (int i = 0; i < NEXP; ++i)
        if (lane == (i & 31)) atomicAdd(&gW[layer * NEXP + i], accW[i]);
    #pragma unroll
    for (int i = 0; i < NEXP / 2; ++i)
        if (lane == i) {
            atomicAdd(&gC[layer * NEXP + i],      (float)(accC[i] & 0xffffu));
            atomicAdd(&gC[layer * NEXP + i + 32], (float)(accC[i] >> 16));
        }
}

// ---------------------------------------------------------------------------
// Stage 2: loss = 0.01 * (E/(T*K)) * sum_{l,e} counts * wsum / T
// WMMA row-sum matmul: D += P_tile(16x32,f16) x ones(32x16,f16); grand-sum of D
// (each rowsum duplicated across 16 cols -> /16). Ones-operand makes the result
// invariant to the exact A lane layout. One wave32, EXEC all ones.
// ---------------------------------------------------------------------------
__global__ __launch_bounds__(32) void bal_stage2(const float* __restrict__ gW,
                                                 const float* __restrict__ gC,
                                                 float* __restrict__ out) {
    const int ln   = threadIdx.x;
    const int half = ln >> 4;     // K-half select per 16-bit A layout
    const int m    = ln & 15;     // row within tile

    v16h ones;
    #pragma unroll
    for (int k = 0; k < 16; ++k) ones[k] = (_Float16)1.0f;

    v8f c = {};
    #pragma unroll
    for (int mt = 0; mt < 2; ++mt) {        // layer tiles 0-15 / 16-31
        #pragma unroll
        for (int kt = 0; kt < 2; ++kt) {    // expert tiles 0-31 / 32-63
            const int l = mt * 16 + m;
            v16h a;
            #pragma unroll
            for (int j = 0; j < 8; ++j) {   // 16-bit A 16x32 VGPR layout
                const int K  = (j < 4) ? (2 * j + 8 * half)
                                       : (2 * (j - 4) + 8 * half + 16);
                const int e0 = kt * 32 + K;
                float p0 = gC[l * 64 + e0]     * gW[l * 64 + e0]     * (1.0f / 16384.0f);
                float p1 = gC[l * 64 + e0 + 1] * gW[l * 64 + e0 + 1] * (1.0f / 16384.0f);
                a[2 * j]     = (_Float16)p0;
                a[2 * j + 1] = (_Float16)p1;
            }
            c = __builtin_amdgcn_wmma_f32_16x16x32_f16(false, a, false, ones,
                                                       (short)0, c, false, false);
        }
    }

    float ssum = 0.0f;
    #pragma unroll
    for (int r = 0; r < 8; ++r) ssum += c[r];
    #pragma unroll
    for (int d = 1; d < 32; d <<= 1) ssum += __shfl_xor(ssum, d, 32);

    if (ln == 0) {
        // 16384 (undo P prescale) * E/(T*K*T) / 16 (dup cols) * 0.01
        const float finalScale =
            (float)((16384.0 * (64.0 / (16384.0 * 8.0 * 16384.0)) / 16.0) * 0.01);
        out[0] = ssum * finalScale;
    }
}

// ---------------------------------------------------------------------------
extern "C" void kernel_launch(void* const* d_in, const int* in_sizes, int n_in,
                              void* d_out, int out_size, void* d_ws, size_t ws_size,
                              hipStream_t stream) {
    (void)in_sizes; (void)n_in; (void)out_size; (void)ws_size;
    const float* logits = (const float*)d_in[0];   // [32,16384,64] f32
    float* gW = (float*)d_ws;                      // [32,64] softmax sums
    float* gC = (float*)d_ws + LAYERS * NEXP;      // [32,64] top-8 counts

    bal_zero  <<<(2 * LAYERS * NEXP + 255) / 256, 256, 0, stream>>>(gW, 2 * LAYERS * NEXP);
    bal_stage1<<<LAYERS * BPL, THREADS, 0, stream>>>(logits, gW, gC);
    bal_stage2<<<1, 32, 0, stream>>>(gW, gC, (float*)d_out);
}
